// TimeAwareGATLayer_14894946582644
// MI455X (gfx1250) — matrix operand order repaired and checked by reading, not compiled, not run on hardware
//
#include <hip/hip_runtime.h>
#include <hip/hip_bf16.h>
#include <math.h>

typedef __attribute__((ext_vector_type(16))) _Float16 v16h;
typedef __attribute__((ext_vector_type(8)))  float    v8f;
typedef __attribute__((ext_vector_type(4)))  float    f32x4;
typedef __attribute__((ext_vector_type(4)))  int      i32x4;
typedef __attribute__((ext_vector_type(4)))  unsigned u32x4;
typedef __attribute__((ext_vector_type(8)))  int      i32x8;

#define B_    4
#define N_    2048
#define INF_  128
#define H_    8
#define HD_   16
#define ALPHA_ 0.2f
#define TDEC_  0.1f
#define NT_   (N_ / 32)          // j-tiles per row-block
#define TS_   36                 // padded LDS row stride (36*4=144B, 16B aligned)

#ifndef USE_TDM
#define USE_TDM __has_builtin(__builtin_amdgcn_tensor_load_to_lds)
#endif

// ---------------------------------------------------------------------------
// TDM: load one 16x32 tile of 4-byte elements (row stride N_) into LDS with
// hardware padding 32dw + 4dw -> LDS row stride 36 elements.  D# per ISA §8.
// This toolchain exposes the 6-arg builtin form:
//   (uint32x4 g0, int32x8 g1, int32x4 g2, int32x4 g3, int32x8 extra, i32 cpol)
// Groups 2/3 and the extra group are unused for a 2D tile -> zeros.
// ---------------------------------------------------------------------------
#if USE_TDM
__device__ __forceinline__ void tdm_tile_load(const void* gptr, unsigned lds_off) {
    unsigned long long ga = (unsigned long long)gptr;
    u32x4 g0;
    g0[0] = 1u;                                              // count=1 (valid user D#)
    g0[1] = lds_off;                                         // lds_addr (bytes)
    g0[2] = (unsigned)ga;                                    // global_addr[31:0]
    g0[3] = ((unsigned)(ga >> 32) & 0x01FFFFFFu)             // global_addr[56:32]
            | 0x80000000u;                                   // type=2 ("image")
    i32x8 g1;
    g1[0] = (2 << 16)            // data_size = 4 bytes
          | (1 << 20)            // pad_enable
          | (4 << 22)            // pad_interval code 4 = 32 DWORDs
          | (3 << 25);           // pad_amount  code 3 = 4 DWORDs
    g1[1] = (int)(N_ << 16);     // tensor_dim0[15:0] = 2048
    g1[2] = (int)(N_ << 16);     // tensor_dim0 hi = 0 | tensor_dim1[15:0] = 2048
    g1[3] = (int)(32u << 16);    // tile_dim0 = 32
    g1[4] = 16;                  // tile_dim1 = 16 (tile_dim2 = 0)
    g1[5] = N_;                  // tensor_dim0_stride = 2048
    g1[6] = 0;
    g1[7] = 0;
    i32x4 z4 = {0, 0, 0, 0};     // groups 2/3: dims 2..4 unused
    i32x8 z8 = {0, 0, 0, 0, 0, 0, 0, 0};
    __builtin_amdgcn_tensor_load_to_lds(g0, g1, z4, z4, z8, 0);
    asm volatile("" ::: "memory");   // LDS is written behind the compiler's back
}
#endif

// ---------------------------------------------------------------------------
// Kernel 0: init scratch scalar (max accumulator, float bits as uint)
// ---------------------------------------------------------------------------
__global__ void ws_init_kernel(unsigned* w) {
    if (threadIdx.x == 0 && blockIdx.x == 0) w[0] = 0u;
}

// ---------------------------------------------------------------------------
// Kernel 1: global max over time_matrix (values >= 0 -> uint bit-compare ==
// float compare; atomicMax order-independent -> deterministic). b128 loads.
// ---------------------------------------------------------------------------
__global__ void __launch_bounds__(256)
time_max_kernel(const float* __restrict__ t, long long n4, unsigned* __restrict__ outbits) {
    float m = 0.f;
    const f32x4* tv = (const f32x4*)t;
    for (long long i = (long long)blockIdx.x * 256 + threadIdx.x; i < n4;
         i += (long long)gridDim.x * 256) {
        f32x4 q = tv[i];
        m = fmaxf(m, fmaxf(fmaxf(q[0], q[1]), fmaxf(q[2], q[3])));
    }
    #pragma unroll
    for (int off = 16; off > 0; off >>= 1)
        m = fmaxf(m, __shfl_down(m, off, 32));
    if ((threadIdx.x & 31) == 0)
        atomicMax(outbits, __float_as_uint(m));
}

// ---------------------------------------------------------------------------
// Kernel 2: h = einsum('bni,hio->bhno') stored f16 transposed ht[b][h][d][n]
//           (WMMA B-operand = two contiguous 8-half loads per lane), plus ei/ej dots.
// ---------------------------------------------------------------------------
__global__ void __launch_bounds__(128)
gat_proj_kernel(const float* __restrict__ x, const float* __restrict__ W,
                const float* __restrict__ a, float* __restrict__ ei,
                float* __restrict__ ej, _Float16* __restrict__ ht) {
    __shared__ float xs[INF_];
    __shared__ float red[128];

    const int bn  = blockIdx.x;
    const int b   = bn >> 11;
    const int n   = bn & (N_ - 1);
    const int tid = threadIdx.x;
    const int h   = tid >> 4;
    const int d   = tid & 15;

    xs[tid] = x[(long long)bn * INF_ + tid];
    __syncthreads();

    float acc = 0.f;
    #pragma unroll 8
    for (int i = 0; i < INF_; ++i)
        acc = fmaf(xs[i], W[(h * INF_ + i) * HD_ + d], acc);

    ht[(long long)((b * H_ + h) * HD_ + d) * N_ + n] = (_Float16)acc;

    red[tid] = acc * a[h * (2 * HD_) + d];
    __syncthreads();
    if (d == 0) {
        float s = 0.f;
        #pragma unroll
        for (int k = 0; k < HD_; ++k) s += red[(h << 4) + k];
        ei[(b * H_ + h) * N_ + n] = s;
    }
    __syncthreads();
    red[tid] = acc * a[h * (2 * HD_) + HD_ + d];
    __syncthreads();
    if (d == 0) {
        float s = 0.f;
        #pragma unroll
        for (int k = 0; k < HD_; ++k) s += red[(h << 4) + k];
        ej[(b * H_ + h) * N_ + n] = s;
    }
}

// ---------------------------------------------------------------------------
// Kernel 3: flash-style masked time-weighted softmax + (att x h) via WMMA.
// 512 blocks x 256 threads (8 waves; wave w == head w). Double-buffered
// 16x32 adj/time tiles staged by TDM (pad -> stride-36 rows). Two passes
// over j (2nd pass L2-resident: 128 MB adj+time < 192 MB L2).
// ---------------------------------------------------------------------------
__global__ void __launch_bounds__(256)
gat_attn_kernel(const int* __restrict__ adj, const float* __restrict__ timem,
                const float* __restrict__ ei, const float* __restrict__ ej,
                const _Float16* __restrict__ ht,
                const unsigned* __restrict__ maxbits,
                float* __restrict__ out) {
    __shared__ float ejs_all[H_ * N_];        // 64 KB: all heads' ej for this b
    __shared__ float time_s[2][16 * TS_];     // double-buffered tiles
    __shared__ int   adj_s [2][16 * TS_];
    __shared__ float sums[H_][HD_];

    const int blk  = blockIdx.x;
    const int b    = blk >> 7;
    const int i0   = (blk & 127) << 4;
    const int tid  = threadIdx.x;
    const int head = tid >> 5;
    const int lane = tid & 31;
    const int row  = lane & 15;      // A-matrix M / B-matrix col / C col
    const int hi   = lane >> 4;
    const int base = hi << 3;        // per-half-wave K offset (ISA A layout)

    const float c0  = TDEC_ * __uint_as_float(maxbits[0]);
    const float eiv = ei[(b * H_ + head) * N_ + i0 + row];
    const long long abase = (long long)(b * N_ + i0) * N_;
    const _Float16* htrow = ht + (long long)((b * H_ + head) * HD_ + row) * N_;

    // preload all ej for this batch (contiguous H_*N_ floats, L2-resident)
    {
        const f32x4* src = (const f32x4*)(ej + (long long)b * H_ * N_);
        f32x4* dst = (f32x4*)ejs_all;
        #pragma unroll
        for (int i = tid; i < (H_ * N_) / 4; i += 256) dst[i] = src[i];
    }

#if !USE_TDM
    // fallback vectorized staging: threads 0..127 adj, 128..255 time, b128 each
    #define STAGE_TILE(JB, BUF)                                                   \
        {                                                                         \
            const int half = tid >> 7, t = tid & 127;                             \
            const int r = t >> 3, c = (t & 7) << 2;                               \
            const long long g = abase + (long long)r * N_ + (long long)(JB) * 32 + c; \
            if (half == 0) *(i32x4*)&adj_s[BUF][r * TS_ + c]  = *(const i32x4*)(adj + g); \
            else           *(f32x4*)&time_s[BUF][r * TS_ + c] = *(const f32x4*)(timem + g); \
        }
#endif

    // =========================== pass 1: row max ===========================
#if USE_TDM
    if (head == 0) {
        tdm_tile_load(adj + abase,   (unsigned)(size_t)&adj_s[0][0]);
        tdm_tile_load(timem + abase, (unsigned)(size_t)&time_s[0][0]);
        __builtin_amdgcn_s_wait_tensorcnt(0);
    }
#else
    STAGE_TILE(0, 0)
#endif
    __syncthreads();

    float m = -INFINITY;
    for (int jb = 0; jb < NT_; ++jb) {
        const int buf = jb & 1;
#if USE_TDM
        if (head == 0 && jb + 1 < NT_) {
            tdm_tile_load(adj + abase + (jb + 1) * 32,   (unsigned)(size_t)&adj_s[buf ^ 1][0]);
            tdm_tile_load(timem + abase + (jb + 1) * 32, (unsigned)(size_t)&time_s[buf ^ 1][0]);
        }
#else
        if (jb + 1 < NT_) STAGE_TILE(jb + 1, buf ^ 1)
#endif
        const float* trow = &time_s[buf][row * TS_ + base];
        const int*   arow = &adj_s [buf][row * TS_ + base];
        const float* erow = &ejs_all[head * N_ + jb * 32 + base];
        f32x4 tq[4]; i32x4 aq[4]; f32x4 eq[4];
        tq[0] = *(const f32x4*)(trow);      tq[1] = *(const f32x4*)(trow + 4);
        tq[2] = *(const f32x4*)(trow + 16); tq[3] = *(const f32x4*)(trow + 20);
        aq[0] = *(const i32x4*)(arow);      aq[1] = *(const i32x4*)(arow + 4);
        aq[2] = *(const i32x4*)(arow + 16); aq[3] = *(const i32x4*)(arow + 20);
        eq[0] = *(const f32x4*)(erow);      eq[1] = *(const f32x4*)(erow + 4);
        eq[2] = *(const f32x4*)(erow + 16); eq[3] = *(const f32x4*)(erow + 20);
        #pragma unroll
        for (int g = 0; g < 4; ++g)
            #pragma unroll
            for (int s = 0; s < 4; ++s) {
                const float tw = __expf(TDEC_ * tq[g][s] - c0);
                float v = eiv + eq[g][s];
                v = (v > 0.f) ? v : ALPHA_ * v;
                v *= tw;
                m = fmaxf(m, (aq[g][s] != 0) ? v : -INFINITY);   // branchless
            }
#if USE_TDM
        if (head == 0 && jb + 1 < NT_) __builtin_amdgcn_s_wait_tensorcnt(0);
#endif
        __syncthreads();
    }
    m = fmaxf(m, __shfl_xor(m, 16, 32));

    // ================== pass 2: exp-sum + P x V via WMMA ===================
#if USE_TDM
    if (head == 0) {
        tdm_tile_load(adj + abase,   (unsigned)(size_t)&adj_s[0][0]);
        tdm_tile_load(timem + abase, (unsigned)(size_t)&time_s[0][0]);
        __builtin_amdgcn_s_wait_tensorcnt(0);
    }
#else
    STAGE_TILE(0, 0)
#endif
    __syncthreads();

    v8f acc = {};
    float rsum = 0.f;
    v16h bcur = __builtin_shufflevector(*(const _Float16 __attribute__((ext_vector_type(8)))*)(htrow + (hi << 3)), *(const _Float16 __attribute__((ext_vector_type(8)))*)(htrow + 16 + (hi << 3)), 0, 1, 2, 3, 4, 5, 6, 7, 8, 9, 10, 11, 12, 13, 14, 15);   // pipelined B operand
    for (int jb = 0; jb < NT_; ++jb) {
        const int buf = jb & 1;
#if USE_TDM
        if (head == 0 && jb + 1 < NT_) {
            tdm_tile_load(adj + abase + (jb + 1) * 32,   (unsigned)(size_t)&adj_s[buf ^ 1][0]);
            tdm_tile_load(timem + abase + (jb + 1) * 32, (unsigned)(size_t)&time_s[buf ^ 1][0]);
        }
#else
        if (jb + 1 < NT_) STAGE_TILE(jb + 1, buf ^ 1)
#endif
        const v16h bnext = (jb + 1 < NT_)
            ? __builtin_shufflevector(*(const _Float16 __attribute__((ext_vector_type(8)))*)(htrow + (jb + 1) * 32 + (hi << 3)), *(const _Float16 __attribute__((ext_vector_type(8)))*)(htrow + (jb + 1) * 32 + 16 + (hi << 3)), 0, 1, 2, 3, 4, 5, 6, 7, 8, 9, 10, 11, 12, 13, 14, 15) : bcur;

        const float* trow = &time_s[buf][row * TS_ + base];
        const int*   arow = &adj_s [buf][row * TS_ + base];
        const float* erow = &ejs_all[head * N_ + jb * 32 + base];
        f32x4 tq[4]; i32x4 aq[4]; f32x4 eq[4];
        tq[0] = *(const f32x4*)(trow);      tq[1] = *(const f32x4*)(trow + 4);
        tq[2] = *(const f32x4*)(trow + 16); tq[3] = *(const f32x4*)(trow + 20);
        aq[0] = *(const i32x4*)(arow);      aq[1] = *(const i32x4*)(arow + 4);
        aq[2] = *(const i32x4*)(arow + 16); aq[3] = *(const i32x4*)(arow + 20);
        eq[0] = *(const f32x4*)(erow);      eq[1] = *(const f32x4*)(erow + 4);
        eq[2] = *(const f32x4*)(erow + 16); eq[3] = *(const f32x4*)(erow + 20);

        v16h amat;
        #pragma unroll
        for (int g = 0; g < 4; ++g)
            #pragma unroll
            for (int s = 0; s < 4; ++s) {
                const float tw = __expf(TDEC_ * tq[g][s] - c0);
                float v = eiv + eq[g][s];
                v = (v > 0.f) ? v : ALPHA_ * v;
                const float pe = __expf(v * tw - m);
                const float p  = (aq[g][s] != 0) ? pe : 0.f;     // branchless
                rsum += p;
                amat[g * 4 + s] = (_Float16)p;
            }
        acc = __builtin_amdgcn_wmma_f32_16x16x32_f16(
            false, amat, false, bcur, (short)0, acc, false, false);
        bcur = bnext;
#if USE_TDM
        if (head == 0 && jb + 1 < NT_) __builtin_amdgcn_s_wait_tensorcnt(0);
#endif
        __syncthreads();
    }

    rsum += __shfl_xor(rsum, 16, 32);
    if (hi == 0) sums[head][row] = rsum;
    __syncthreads();

    // C/D layout: VGPR r -> row r + 8*hi, column = lane&15
    #pragma unroll
    for (int r = 0; r < 8; ++r) {
        const int orow = r + (hi << 3);
        const float v = acc[r] / sums[head][orow];
        const float o = (v > 0.f) ? v : (expf(v) - 1.f);
        out[(long long)(b * N_ + i0 + orow) * (H_ * HD_) + head * HD_ + row] = o;
    }
}

// ---------------------------------------------------------------------------
extern "C" void kernel_launch(void* const* d_in, const int* in_sizes, int n_in,
                              void* d_out, int out_size, void* d_ws, size_t ws_size,
                              hipStream_t stream) {
    const float* x     = (const float*)d_in[0];  // (B,N,128)
    const int*   adj   = (const int*)  d_in[1];  // (B,N,N)
    const float* timem = (const float*)d_in[2];  // (B,N,N)
    const float* W     = (const float*)d_in[3];  // (H,128,16)
    const float* a     = (const float*)d_in[4];  // (H,32,1)
    float* out = (float*)d_out;                  // (B,N,128)

    char* ws = (char*)d_ws;
    unsigned* maxbits = (unsigned*)ws;                        // 4 B
    float*    ei = (float*)(ws + 64);                         // 256 KB
    float*    ej = (float*)(ws + 64 + 4 * B_ * H_ * N_);      // 256 KB
    _Float16* ht = (_Float16*)(ws + 64 + 8 * B_ * H_ * N_);   // 2 MB

    ws_init_kernel<<<1, 1, 0, stream>>>(maxbits);
    time_max_kernel<<<2048, 256, 0, stream>>>(timem, ((long long)B_ * N_ * N_) / 4, maxbits);
    gat_proj_kernel<<<B_ * N_, 128, 0, stream>>>(x, W, a, ei, ej, ht);
    gat_attn_kernel<<<B_ * (N_ / 16), 256, 0, stream>>>(adj, timem, ei, ej, ht, maxbits, out);
}
